// AR_Back_Step_14267881357761
// MI455X (gfx1250) — compile-verified, matrix operand order checked by reading, not run on hardware
//
#include <hip/hip_runtime.h>
#include <hip/hip_bf16.h>

// ---------------------------------------------------------------------------
// Problem constants (from reference)
// ---------------------------------------------------------------------------
#define HN    1024
#define MEL   80
#define MELP  96          // MEL padded to multiple of 32 for WMMA K
#define ATT   640
#define TXTD  640
#define TSTEPS 600
#define TTXT  128
#define G4    4096        // 4*H
#define KA    1120        // MELP + H   (fused attention-LSTM input)
#define KD0   1664        // H + ATT    (decoder LSTM0 input)

#define NWG   64          // persistent workgroups
#define NTHR  256         // 8 wave32 per WG
#define NWAVES (NWG*8)

typedef __bf16 bf16;
typedef __attribute__((ext_vector_type(16))) __bf16 bf16x16;
typedef __attribute__((ext_vector_type(8)))  __bf16 bf16x8;
typedef __attribute__((ext_vector_type(8)))  float  v8f;
typedef int v4i_ __attribute__((vector_size(16)));   // matches builtin param type

union FragBF { bf16x16 v; bf16x8 h[2]; };

#if __has_builtin(__builtin_amdgcn_global_load_async_to_lds_b128) && \
    __has_builtin(__builtin_amdgcn_s_wait_asynccnt)
#define HAVE_ASYNC_LDS 1
#endif

// ---------------------------------------------------------------------------
// Workspace layout (byte offsets, 256B aligned)
// ---------------------------------------------------------------------------
constexpr size_t al256(size_t x) { return (x + 255) & ~(size_t)255; }
constexpr size_t OFF_BAR = 0;                                      // 2 x u32
constexpr size_t OFF_C   = 256;                                    // c   [2][1024] f32
constexpr size_t OFF_C0  = al256(OFF_C   + 2*HN*4);                // c0  [2][1024] f32
constexpr size_t OFF_C1  = al256(OFF_C0  + 2*HN*4);                // c1  [2][1024] f32
constexpr size_t OFF_GA  = al256(OFF_C1  + 2*HN*4);                // attn-lstm gates f32[4096]
constexpr size_t OFF_G0  = al256(OFF_GA  + G4*4);                  // lstm0 gates
constexpr size_t OFF_G1  = al256(OFF_G0  + G4*4);                  // lstm1 gates
constexpr size_t OFF_BAS = al256(OFF_G1  + G4*4);                  // bih_a+bhh_a
constexpr size_t OFF_B0S = al256(OFF_BAS + G4*4);
constexpr size_t OFF_B1S = al256(OFF_B0S + G4*4);
constexpr size_t OFF_Q   = al256(OFF_B1S + G4*4);                  // q f32[640]
constexpr size_t OFF_XA  = al256(OFF_Q   + ATT*4);                 // [out_prev(96) | h] bf16[1120]
constexpr size_t OFF_DEC = al256(OFF_XA  + KA*2);                  // dec_in bf16[1664]
constexpr size_t OFF_H0B = al256(OFF_DEC + KD0*2);                 // h0 bf16[1024]
constexpr size_t OFF_H1B = al256(OFF_H0B + HN*2);                  // h1 bf16[1024]
constexpr size_t OFF_A0B = al256(OFF_H1B + HN*2);                  // tanh(D0w h1+b) bf16
constexpr size_t OFF_DB  = al256(OFF_A0B + HN*2);                  // d bf16[1024]
constexpr size_t OFF_KM  = al256(OFF_DB  + HN*2);                  // K f32[128*640]
constexpr size_t OFF_VM  = al256(OFF_KM  + (size_t)TTXT*ATT*4);    // V f32[128*640]
constexpr size_t OFF_WA  = al256(OFF_VM  + (size_t)TTXT*ATT*4);    // [Wih_a|Whh_a] bf16 4096x1120
constexpr size_t OFF_WQ  = al256(OFF_WA  + (size_t)G4*KA*2);       // Wq bf16 640x1024
constexpr size_t OFF_WI0 = al256(OFF_WQ  + (size_t)ATT*HN*2);      // Wih0 bf16 4096x1664
constexpr size_t OFF_WH0 = al256(OFF_WI0 + (size_t)G4*KD0*2);      // Whh0 bf16 4096x1024
constexpr size_t OFF_WI1 = al256(OFF_WH0 + (size_t)G4*HN*2);
constexpr size_t OFF_WH1 = al256(OFF_WI1 + (size_t)G4*HN*2);
constexpr size_t OFF_D0W = al256(OFF_WH1 + (size_t)G4*HN*2);       // 1024x1024 bf16
constexpr size_t OFF_D1W = al256(OFF_D0W + (size_t)HN*HN*2);
constexpr size_t OFF_CW  = al256(OFF_D1W + (size_t)HN*HN*2);       // conv_w bf16 160x1024

// ---------------------------------------------------------------------------
// Helpers
// ---------------------------------------------------------------------------
__device__ __forceinline__ float sigf(float x) { return 1.0f / (1.0f + __expf(-x)); }
__device__ __forceinline__ float ftanh(float x) {
    float e = __expf(2.0f * x);
    return 1.0f - 2.0f / (e + 1.0f);
}

// One 16-row GEMV tile: y[0..15] = Wt(16xK) @ x, using WMMA with x replicated
// across all 16 B-columns. A/B fragment k-permutation per ISA 16-bit layout:
// lanes 0-15 hold K {0..7,16..23} of each 32-chunk, lanes 16-31 hold {8..15,24..31}.
__device__ __forceinline__ v8f wmma_gemv_tile(const bf16* __restrict__ Wt,
                                              const bf16* __restrict__ x,
                                              int K, int lane)
{
    const int row = lane & 15;
    const int klo = (lane >> 4) << 3;           // 0 or 8
    const bf16* wr = Wt + (size_t)row * K + klo;
    const bf16* xr = x + klo;
    v8f acc = {};
    for (int k = 0; k < K; k += 32) {
        FragBF a, b;
        a.h[0] = *(const bf16x8*)(wr + k);
        a.h[1] = *(const bf16x8*)(wr + k + 16);
        b.h[0] = *(const bf16x8*)(xr + k);
        b.h[1] = *(const bf16x8*)(xr + k + 16);
        __builtin_prefetch(wr + k + 512, 0, 3);   // near-cache prefetch of next chunks
        acc = __builtin_amdgcn_wmma_f32_16x16x32_bf16(
                  false, a.v, false, b.v, (short)0, acc, false, false);
    }
    return acc;   // lane 0 holds rows 0..7 in acc[0..7]; lane 16 rows 8..15
}

// Issue an async (or plain) copy of a small activation vector global -> LDS.
// n16 = number of 16-byte chunks. Caller must s_wait_asynccnt + __syncthreads
// (use stage_wait()) before consuming.
__device__ __forceinline__ void stage_issue(const bf16* __restrict__ g,
                                            bf16* __restrict__ l, int n16)
{
#ifdef HAVE_ASYNC_LDS
    for (int i = threadIdx.x; i < n16; i += NTHR) {
        __builtin_amdgcn_global_load_async_to_lds_b128(
            (__attribute__((address_space(1))) v4i_*)
                (unsigned long long)((const char*)g + (size_t)i * 16),
            (__attribute__((address_space(3))) v4i_*)
                (unsigned)(unsigned long long)((char*)l + (size_t)i * 16),
            0, 0);
    }
#else
    for (int i = threadIdx.x; i < n16; i += NTHR) {
        ((int4*)l)[i] = ((const int4*)g)[i];
    }
#endif
}

__device__ __forceinline__ void stage_wait()
{
#ifdef HAVE_ASYNC_LDS
    __builtin_amdgcn_s_wait_asynccnt(0);
#endif
    __syncthreads();
}

__device__ __forceinline__ void grid_barrier(unsigned* bar)
{
    __syncthreads();
    if (threadIdx.x == 0) {
        unsigned* cnt = bar;
        unsigned* gen = bar + 1;
        __threadfence();
        unsigned g = __hip_atomic_load(gen, __ATOMIC_ACQUIRE, __HIP_MEMORY_SCOPE_AGENT);
        unsigned arrived = __hip_atomic_fetch_add(cnt, 1u, __ATOMIC_ACQ_REL,
                                                  __HIP_MEMORY_SCOPE_AGENT);
        if (arrived == NWG - 1) {
            __hip_atomic_store(cnt, 0u, __ATOMIC_RELAXED, __HIP_MEMORY_SCOPE_AGENT);
            __hip_atomic_fetch_add(gen, 1u, __ATOMIC_ACQ_REL, __HIP_MEMORY_SCOPE_AGENT);
        } else {
            while (__hip_atomic_load(gen, __ATOMIC_ACQUIRE,
                                     __HIP_MEMORY_SCOPE_AGENT) == g) {
                __builtin_amdgcn_s_sleep(1);
            }
        }
    }
    __syncthreads();
}

__device__ __forceinline__ void conv_to_bf16(const float* __restrict__ src,
                                             bf16* __restrict__ dst,
                                             size_t n, size_t gt, size_t st)
{
    for (size_t i = gt; i < n; i += st) dst[i] = (bf16)src[i];
}

// ---------------------------------------------------------------------------
// Kernel 1: init barriers/state, fold biases, convert weights to bf16, K/V
// ---------------------------------------------------------------------------
__global__ void __launch_bounds__(NTHR) ar_prep_kernel(
    const float* __restrict__ Wih_a, const float* __restrict__ Whh_a,
    const float* __restrict__ bih_a, const float* __restrict__ bhh_a,
    const float* __restrict__ Wq,    const float* __restrict__ Wk,
    const float* __restrict__ Wv,
    const float* __restrict__ Wih0,  const float* __restrict__ Whh0,
    const float* __restrict__ bih0,  const float* __restrict__ bhh0,
    const float* __restrict__ Wih1,  const float* __restrict__ Whh1,
    const float* __restrict__ bih1,  const float* __restrict__ bhh1,
    const float* __restrict__ D0w,   const float* __restrict__ D1w,
    const float* __restrict__ convw, const float* __restrict__ text,
    char* __restrict__ ws)
{
    const size_t gt = (size_t)blockIdx.x * blockDim.x + threadIdx.x;
    const size_t st = (size_t)gridDim.x * blockDim.x;

    unsigned* bar = (unsigned*)(ws + OFF_BAR);
    float* Cf  = (float*)(ws + OFF_C);
    float* C0f = (float*)(ws + OFF_C0);
    float* C1f = (float*)(ws + OFF_C1);
    float* BAS = (float*)(ws + OFF_BAS);
    float* B0S = (float*)(ws + OFF_B0S);
    float* B1S = (float*)(ws + OFF_B1S);
    float* KM  = (float*)(ws + OFF_KM);
    float* VM  = (float*)(ws + OFF_VM);
    bf16* XA   = (bf16*)(ws + OFF_XA);
    bf16* H0B  = (bf16*)(ws + OFF_H0B);
    bf16* H1B  = (bf16*)(ws + OFF_H1B);
    bf16* WAb  = (bf16*)(ws + OFF_WA);
    bf16* WQb  = (bf16*)(ws + OFF_WQ);
    bf16* WI0b = (bf16*)(ws + OFF_WI0);
    bf16* WH0b = (bf16*)(ws + OFF_WH0);
    bf16* WI1b = (bf16*)(ws + OFF_WI1);
    bf16* WH1b = (bf16*)(ws + OFF_WH1);
    bf16* D0Wb = (bf16*)(ws + OFF_D0W);
    bf16* D1Wb = (bf16*)(ws + OFF_D1W);
    bf16* CWb  = (bf16*)(ws + OFF_CW);

    if (gt == 0) { bar[0] = 0u; bar[1] = 0u; }
    for (size_t i = gt; i < 2 * HN; i += st) { Cf[i] = 0.f; C0f[i] = 0.f; C1f[i] = 0.f; }
    for (size_t i = gt; i < HN; i += st) { H0B[i] = (bf16)0.f; H1B[i] = (bf16)0.f; }
    for (size_t i = gt; i < KA; i += st) XA[i] = (bf16)0.f;
    for (size_t i = gt; i < G4; i += st) {
        BAS[i] = bih_a[i] + bhh_a[i];
        B0S[i] = bih0[i]  + bhh0[i];
        B1S[i] = bih1[i]  + bhh1[i];
    }
    // Fused + padded attention-LSTM weight: [Wih_a (80, pad 16 zero) | Whh_a]
    for (size_t i = gt; i < (size_t)G4 * KA; i += st) {
        size_t r = i / KA, k = i % KA;
        float v = (k < MEL)  ? Wih_a[r * MEL + k]
                : (k < MELP) ? 0.f
                             : Whh_a[r * HN + (k - MELP)];
        WAb[i] = (bf16)v;
    }
    conv_to_bf16(Wq,    WQb,  (size_t)ATT * HN,  gt, st);
    conv_to_bf16(Wih0,  WI0b, (size_t)G4 * KD0,  gt, st);
    conv_to_bf16(Whh0,  WH0b, (size_t)G4 * HN,   gt, st);
    conv_to_bf16(Wih1,  WI1b, (size_t)G4 * HN,   gt, st);
    conv_to_bf16(Whh1,  WH1b, (size_t)G4 * HN,   gt, st);
    conv_to_bf16(D0w,   D0Wb, (size_t)HN * HN,   gt, st);
    conv_to_bf16(D1w,   D1Wb, (size_t)HN * HN,   gt, st);
    conv_to_bf16(convw, CWb,  (size_t)2*MEL*HN,  gt, st);
    // K = text @ Wk^T ; V = text @ Wv^T   (done once)
    for (size_t i = gt; i < (size_t)TTXT * ATT; i += st) {
        size_t t = i / ATT, a = i % ATT;
        const float* tx = text + t * TXTD;
        const float* wk = Wk + a * TXTD;
        const float* wv = Wv + a * TXTD;
        float sk = 0.f, sv = 0.f;
        for (int j = 0; j < TXTD; ++j) { float x = tx[j]; sk += x * wk[j]; sv += x * wv[j]; }
        KM[i] = sk; VM[i] = sv;
    }
}

// ---------------------------------------------------------------------------
// Kernel 2: persistent 600-step autoregressive loop
// ---------------------------------------------------------------------------
__global__ void __launch_bounds__(NTHR, 1) ar_step_kernel(
    const float* __restrict__ RES,  const float* __restrict__ VW,
    const float* __restrict__ D0B,  const float* __restrict__ D1B,
    const float* __restrict__ CB,   const float* __restrict__ GW,
    const float* __restrict__ GB,   float* __restrict__ OUT,
    char* __restrict__ ws)
{
    unsigned* bar = (unsigned*)(ws + OFF_BAR);
    float* Cf  = (float*)(ws + OFF_C);
    float* C0f = (float*)(ws + OFF_C0);
    float* C1f = (float*)(ws + OFF_C1);
    float* GAf = (float*)(ws + OFF_GA);
    float* G0f = (float*)(ws + OFF_G0);
    float* G1f = (float*)(ws + OFF_G1);
    float* BAS = (float*)(ws + OFF_BAS);
    float* B0S = (float*)(ws + OFF_B0S);
    float* B1S = (float*)(ws + OFF_B1S);
    float* Qf  = (float*)(ws + OFF_Q);
    float* KM  = (float*)(ws + OFF_KM);
    float* VM  = (float*)(ws + OFF_VM);
    bf16* XA   = (bf16*)(ws + OFF_XA);
    bf16* DECIN= (bf16*)(ws + OFF_DEC);
    bf16* H0B  = (bf16*)(ws + OFF_H0B);
    bf16* H1B  = (bf16*)(ws + OFF_H1B);
    bf16* A0B  = (bf16*)(ws + OFF_A0B);
    bf16* DBv  = (bf16*)(ws + OFF_DB);
    const bf16* WAb  = (const bf16*)(ws + OFF_WA);
    const bf16* WQb  = (const bf16*)(ws + OFF_WQ);
    const bf16* WI0b = (const bf16*)(ws + OFF_WI0);
    const bf16* WH0b = (const bf16*)(ws + OFF_WH0);
    const bf16* WI1b = (const bf16*)(ws + OFF_WI1);
    const bf16* WH1b = (const bf16*)(ws + OFF_WH1);
    const bf16* D0Wb = (const bf16*)(ws + OFF_D0W);
    const bf16* D1Wb = (const bf16*)(ws + OFF_D1W);
    const bf16* CWb  = (const bf16*)(ws + OFF_CW);

    const int tid   = threadIdx.x;
    const int lane  = tid & 31;
    const int wid   = tid >> 5;
    const int gwave = blockIdx.x * 8 + wid;

    __shared__ float sh_h[HN];      // attn-LSTM h (f32, kept through stage 3)
    __shared__ bf16  sh_bf[HN];     // bf16 vector computed locally (h / h0 / h1)
    __shared__ bf16  sh_x[KD0];     // staged copy of a global activation vector
    __shared__ bf16  sh_y[HN];      // second staged vector (h1_prev in stage 2)
    __shared__ float s_ctx[ATT];
    __shared__ float s_sc[TTXT];
    __shared__ float s_red[NTHR];
    __shared__ float s_dec[2 * MEL];

    for (int step = 0; step < TSTEPS; ++step) {
        const int p = step & 1;

        // ---- Stage 1: attention-LSTM gates = [Wih_a|Whh_a] @ [out_prev|h] + bias
        stage_issue(XA, sh_x, KA * 2 / 16);
        stage_wait();
        for (int tile = gwave; tile < 256; tile += NWAVES) {
            v8f acc = wmma_gemv_tile(WAb + (size_t)tile * 16 * KA, sh_x, KA, lane);
            if ((lane & 15) == 0) {
                int mb = tile * 16 + ((lane >> 4) << 3);
                #pragma unroll
                for (int v = 0; v < 8; ++v) GAf[mb + v] = acc[v] + BAS[mb + v];
            }
        }
        grid_barrier(bar);

        // ---- Stage 2: h,c update (redundant per-WG), overlapped with async
        //      staging of h0_prev/h1_prev; then q = Wq@h, g0 = Whh0@h0_prev + b,
        //      g1 = Whh1@h1_prev + b
        {
            stage_issue(H0B, sh_x, HN * 2 / 16);   // async: h0_prev -> LDS
            stage_issue(H1B, sh_y, HN * 2 / 16);   // async: h1_prev -> LDS
            const float* cp = Cf + p * HN;
            float*       cn = Cf + (p ^ 1) * HN;
            for (int j = tid; j < HN; j += NTHR) {
                float gi = GAf[j], gf = GAf[j + HN], gg = GAf[j + 2*HN], go = GAf[j + 3*HN];
                float c2 = sigf(gf) * cp[j] + sigf(gi) * ftanh(gg);
                float hh = sigf(go) * ftanh(c2);
                sh_h[j]  = hh;
                sh_bf[j] = (bf16)hh;
                if (blockIdx.x == 0) {
                    cn[j] = c2;
                    XA[MELP + j] = (bf16)hh;   // for next step's stage 1
                    DECIN[j]     = (bf16)hh;   // for this step's stage 4
                }
            }
            stage_wait();
            for (int task = gwave; task < 552; task += NWAVES) {
                if (task < 40) {               // q tiles
                    v8f acc = wmma_gemv_tile(WQb + (size_t)task * 16 * HN, sh_bf, HN, lane);
                    if ((lane & 15) == 0) {
                        int mb = task * 16 + ((lane >> 4) << 3);
                        #pragma unroll
                        for (int v = 0; v < 8; ++v) Qf[mb + v] = acc[v];
                    }
                } else if (task < 296) {       // g0 recurrent part
                    int tl = task - 40;
                    v8f acc = wmma_gemv_tile(WH0b + (size_t)tl * 16 * HN, sh_x, HN, lane);
                    if ((lane & 15) == 0) {
                        int mb = tl * 16 + ((lane >> 4) << 3);
                        #pragma unroll
                        for (int v = 0; v < 8; ++v) G0f[mb + v] = acc[v] + B0S[mb + v];
                    }
                } else {                       // g1 recurrent part
                    int tl = task - 296;
                    v8f acc = wmma_gemv_tile(WH1b + (size_t)tl * 16 * HN, sh_y, HN, lane);
                    if ((lane & 15) == 0) {
                        int mb = tl * 16 + ((lane >> 4) << 3);
                        #pragma unroll
                        for (int v = 0; v < 8; ++v) G1f[mb + v] = acc[v] + B1S[mb + v];
                    }
                }
            }
        }
        grid_barrier(bar);

        // ---- Stage 3 (WG0): scores/softmax/ctx/gate
        if (blockIdx.x == 0) {
            for (int t2 = tid; t2 < TTXT; t2 += NTHR) {
                const float* Kr = KM + t2 * ATT;
                float s = 0.f;
                for (int a = 0; a < ATT; ++a) s += VW[a] * ftanh(Qf[a] + Kr[a]);
                s_sc[t2] = s;
            }
            __syncthreads();
            if (tid == 0) {
                float mx = s_sc[0];
                for (int t2 = 1; t2 < TTXT; ++t2) mx = fmaxf(mx, s_sc[t2]);
                float sum = 0.f;
                for (int t2 = 0; t2 < TTXT; ++t2) { float e = __expf(s_sc[t2] - mx); s_sc[t2] = e; sum += e; }
                float inv = 1.f / sum;
                for (int t2 = 0; t2 < TTXT; ++t2) s_sc[t2] *= inv;
            }
            __syncthreads();
            for (int a = tid; a < ATT; a += NTHR) {
                float acc = 0.f;
                for (int t2 = 0; t2 < TTXT; ++t2) acc += s_sc[t2] * VM[t2 * ATT + a];
                s_ctx[a] = acc;
                DECIN[HN + a] = (bf16)acc;
            }
            __syncthreads();
            float part = 0.f;
            for (int j = tid; j < KD0; j += NTHR)
                part += GW[j] * (j < HN ? sh_h[j] : s_ctx[j - HN]);
            s_red[tid] = part;
            __syncthreads();
            for (int off = NTHR / 2; off > 0; off >>= 1) {
                if (tid < off) s_red[tid] += s_red[tid + off];
                __syncthreads();
            }
            if (tid == 0) OUT[TSTEPS * MEL + step] = sigf(s_red[0] + GB[0]);
        }
        grid_barrier(bar);

        // ---- Stage 4: g0 += Wih0 @ dec_in
        stage_issue(DECIN, sh_x, KD0 * 2 / 16);
        stage_wait();
        for (int tile = gwave; tile < 256; tile += NWAVES) {
            v8f acc = wmma_gemv_tile(WI0b + (size_t)tile * 16 * KD0, sh_x, KD0, lane);
            if ((lane & 15) == 0) {
                int mb = tile * 16 + ((lane >> 4) << 3);
                #pragma unroll
                for (int v = 0; v < 8; ++v) G0f[mb + v] += acc[v];
            }
        }
        grid_barrier(bar);

        // ---- Stage 5: h0,c0 update; g1 += Wih1 @ h0
        {
            const float* cp = C0f + p * HN;
            float*       cn = C0f + (p ^ 1) * HN;
            for (int j = tid; j < HN; j += NTHR) {
                float gi = G0f[j], gf = G0f[j + HN], gg = G0f[j + 2*HN], go = G0f[j + 3*HN];
                float c2 = sigf(gf) * cp[j] + sigf(gi) * ftanh(gg);
                float hh = sigf(go) * ftanh(c2);
                sh_bf[j] = (bf16)hh;
                if (blockIdx.x == 0) { cn[j] = c2; H0B[j] = (bf16)hh; }
            }
            __syncthreads();
            for (int tile = gwave; tile < 256; tile += NWAVES) {
                v8f acc = wmma_gemv_tile(WI1b + (size_t)tile * 16 * HN, sh_bf, HN, lane);
                if ((lane & 15) == 0) {
                    int mb = tile * 16 + ((lane >> 4) << 3);
                    #pragma unroll
                    for (int v = 0; v < 8; ++v) G1f[mb + v] += acc[v];
                }
            }
        }
        grid_barrier(bar);

        // ---- Stage 6: h1,c1 update; a0 = tanh(D0w@h1 + D0b)
        {
            const float* cp = C1f + p * HN;
            float*       cn = C1f + (p ^ 1) * HN;
            for (int j = tid; j < HN; j += NTHR) {
                float gi = G1f[j], gf = G1f[j + HN], gg = G1f[j + 2*HN], go = G1f[j + 3*HN];
                float c2 = sigf(gf) * cp[j] + sigf(gi) * ftanh(gg);
                float hh = sigf(go) * ftanh(c2);
                sh_bf[j] = (bf16)hh;
                if (blockIdx.x == 0) { cn[j] = c2; H1B[j] = (bf16)hh; }
            }
            __syncthreads();
            for (int tile = gwave; tile < 64; tile += NWAVES) {
                v8f acc = wmma_gemv_tile(D0Wb + (size_t)tile * 16 * HN, sh_bf, HN, lane);
                if ((lane & 15) == 0) {
                    int mb = tile * 16 + ((lane >> 4) << 3);
                    #pragma unroll
                    for (int v = 0; v < 8; ++v) A0B[mb + v] = (bf16)ftanh(acc[v] + D0B[mb + v]);
                }
            }
        }
        grid_barrier(bar);

        // ---- Stage 7: d = tanh(D1w@a0 + D1b)
        stage_issue(A0B, sh_x, HN * 2 / 16);
        stage_wait();
        for (int tile = gwave; tile < 64; tile += NWAVES) {
            v8f acc = wmma_gemv_tile(D1Wb + (size_t)tile * 16 * HN, sh_x, HN, lane);
            if ((lane & 15) == 0) {
                int mb = tile * 16 + ((lane >> 4) << 3);
                #pragma unroll
                for (int v = 0; v < 8; ++v) DBv[mb + v] = (bf16)ftanh(acc[v] + D1B[mb + v]);
            }
        }
        grid_barrier(bar);

        // ---- Stage 8 (WG0): dec_out = conv_w@d + conv_b; emit output
        if (blockIdx.x == 0) {
            stage_issue(DBv, sh_x, HN * 2 / 16);
            stage_wait();
            for (int tile = wid; tile < 10; tile += 8) {
                v8f acc = wmma_gemv_tile(CWb + (size_t)tile * 16 * HN, sh_x, HN, lane);
                if ((lane & 15) == 0) {
                    int mb = tile * 16 + ((lane >> 4) << 3);
                    #pragma unroll
                    for (int v = 0; v < 8; ++v) s_dec[mb + v] = acc[v] + CB[mb + v];
                }
            }
            __syncthreads();
            const int ridx = TSTEPS - 1 - step;
            for (int m = tid; m < MEL; m += NTHR) {
                float ls = s_dec[m], bb = s_dec[MEL + m];
                float o  = (RES[ridx * MEL + m] - bb) * __expf(-ls);
                OUT[ridx * MEL + m] = o;      // total[ridx]
                XA[m] = (bf16)o;              // out_prev for next step
            }
        }
        grid_barrier(bar);
    }
}

// ---------------------------------------------------------------------------
// Host launcher
// ---------------------------------------------------------------------------
extern "C" void kernel_launch(void* const* d_in, const int* in_sizes, int n_in,
                              void* d_out, int out_size, void* d_ws, size_t ws_size,
                              hipStream_t stream)
{
    (void)in_sizes; (void)n_in; (void)out_size; (void)ws_size;
    const float* residual = (const float*)d_in[0];
    const float* text     = (const float*)d_in[1];
    const float* Wih_a    = (const float*)d_in[2];
    const float* Whh_a    = (const float*)d_in[3];
    const float* bih_a    = (const float*)d_in[4];
    const float* bhh_a    = (const float*)d_in[5];
    const float* Wq       = (const float*)d_in[6];
    const float* Wk       = (const float*)d_in[7];
    const float* Wv       = (const float*)d_in[8];
    const float* v_w      = (const float*)d_in[9];
    const float* Wih0     = (const float*)d_in[10];
    const float* Whh0     = (const float*)d_in[11];
    const float* bih0     = (const float*)d_in[12];
    const float* bhh0     = (const float*)d_in[13];
    const float* Wih1     = (const float*)d_in[14];
    const float* Whh1     = (const float*)d_in[15];
    const float* bih1     = (const float*)d_in[16];
    const float* bhh1     = (const float*)d_in[17];
    const float* D0w      = (const float*)d_in[18];
    const float* D0b      = (const float*)d_in[19];
    const float* D1w      = (const float*)d_in[20];
    const float* D1b      = (const float*)d_in[21];
    const float* conv_w   = (const float*)d_in[22];
    const float* conv_b   = (const float*)d_in[23];
    const float* gate_w   = (const float*)d_in[24];
    const float* gate_b   = (const float*)d_in[25];
    char* ws = (char*)d_ws;

    ar_prep_kernel<<<512, NTHR, 0, stream>>>(
        Wih_a, Whh_a, bih_a, bhh_a, Wq, Wk, Wv,
        Wih0, Whh0, bih0, bhh0, Wih1, Whh1, bih1, bhh1,
        D0w, D1w, conv_w, text, ws);

    ar_step_kernel<<<NWG, NTHR, 0, stream>>>(
        residual, v_w, D0b, D1b, conv_b, gate_w, gate_b, (float*)d_out, ws);
}
